// CodePredictorAttention_75488345194670
// MI455X (gfx1250) — compile-verified, matrix operand order checked by reading, not run on hardware
//
#include <hip/hip_runtime.h>
#include <hip/hip_bf16.h>

// ---------------------------------------------------------------------------
// CodePredictorAttention for MI455X (gfx1250, wave32, WMMA bf16 16x16x32)
// ---------------------------------------------------------------------------
#define B_   2
#define Q_   2048
#define HID_ 2048
#define HQ_  16
#define HKV_ 4
#define D_   128
#define QKVN (HQ_*D_ + 2*HKV_*D_)   // 3072
#define NTOK (B_*Q_)                 // 4096

typedef __attribute__((ext_vector_type(16))) __bf16 v16bf;
typedef __attribute__((ext_vector_type(8)))  float  v8f;
typedef int i4vs __attribute__((vector_size(16)));   // matches builtin param type

// gfx1250 async global->LDS path (guarded: falls back to reg staging if the
// toolchain doesn't expose the builtin; host pass always takes the fallback).
#if defined(__HIP_DEVICE_COMPILE__) && __has_builtin(__builtin_amdgcn_global_load_async_to_lds_b128)
#define HAS_ASYNC_LDS 1
#else
#define HAS_ASYNC_LDS 0
#endif

__device__ __forceinline__ void wait_async_lds() {
#if HAS_ASYNC_LDS
#if __has_builtin(__builtin_amdgcn_s_wait_asynccnt)
    __builtin_amdgcn_s_wait_asynccnt(0);
#else
    asm volatile("s_wait_asynccnt 0" ::: "memory");
#endif
#endif
}

__device__ __forceinline__ __bf16 f2bf(float f) {
    unsigned u = __builtin_bit_cast(unsigned, f);
    unsigned r = u + 0x7FFFu + ((u >> 16) & 1u);   // round-to-nearest-even
    unsigned short s = (unsigned short)(r >> 16);
    return __builtin_bit_cast(__bf16, s);
}
__device__ __forceinline__ __bf16 to_bf16(float f)          { return f2bf(f); }
__device__ __forceinline__ __bf16 to_bf16(unsigned short u) { return __builtin_bit_cast(__bf16, u); }

// K-index inside a 32-wide contraction chunk for 16-bit A/B fragments
// (ISA 7.12.2: lanes 0-15 hold K 0..7 then 16..23; lanes 16-31 hold 8..15 then 24..31)
__device__ __forceinline__ int fragk(int lane, int h) {
    return ((h >> 3) << 4) | (((lane >> 4) & 1) << 3) | (h & 7);
}

// ---------------------------------------------------------------------------
// Generic GEMM: C[M][N] = A[M][K] * B[N][K]^T   (B row-major N x K)
// A element type AT (float or bf16-as-ushort); B is float. f32 accumulate.
// 128x128 block tile, 256 threads = 8 waves (4x2), BK = 32, double-buffered
// LDS with register prefetch of the next K-tile.
// ---------------------------------------------------------------------------
template <typename AT>
__global__ __launch_bounds__(256)
void gemm_wmma_bf16(const AT* __restrict__ A, const float* __restrict__ Bm,
                    float* __restrict__ C, int M, int N, int K)
{
    const int LDSS = 40;                 // 32 + 8 pad (halves)
    __shared__ __bf16 As[2][128 * LDSS];
    __shared__ __bf16 Bs[2][128 * LDSS];

    const int tid  = threadIdx.x;
    const int wid  = tid >> 5;
    const int lane = tid & 31;
    const int l15  = lane & 15;
    const int hi   = (lane >> 4) & 1;
    const int wm   = wid >> 1;           // 0..3  -> 32 rows each
    const int wn   = wid & 1;            // 0..1  -> 64 cols each
    const int bm   = blockIdx.y * 128;
    const int bn   = blockIdx.x * 128;

    const int row = tid >> 1;            // 0..127
    const int cb  = (tid & 1) * 16;      // 0 or 16

    v8f acc[2][4];
#pragma unroll
    for (int i = 0; i < 2; ++i)
#pragma unroll
        for (int j = 0; j < 4; ++j) acc[i][j] = (v8f){};

    AT    aReg[16];
    float bReg[16];
    auto loadTile = [&](int kb) {
        const AT*    ap = A  + (size_t)(bm + row) * K + kb + cb;
        const float* bp = Bm + (size_t)(bn + row) * K + kb + cb;
#pragma unroll
        for (int c = 0; c < 16; ++c) { aReg[c] = ap[c]; bReg[c] = bp[c]; }
    };
    auto storeTile = [&](int buf) {
#pragma unroll
        for (int c = 0; c < 16; ++c) {
            As[buf][row * LDSS + cb + c] = to_bf16(aReg[c]);
            Bs[buf][row * LDSS + cb + c] = f2bf(bReg[c]);
        }
    };

    loadTile(0);
    storeTile(0);
    __syncthreads();

    int cur = 0;
    for (int kb = 0; kb < K; kb += 32) {
        const bool has_next = (kb + 32 < K);
        if (has_next) loadTile(kb + 32);          // global prefetch overlaps WMMA

        v16bf af[2], bf[4];
#pragma unroll
        for (int i = 0; i < 2; ++i) {
            const int m = wm * 32 + i * 16 + l15;
#pragma unroll
            for (int h = 0; h < 16; ++h)
                af[i][h] = As[cur][m * LDSS + fragk(lane, h)];
        }
#pragma unroll
        for (int j = 0; j < 4; ++j) {
            const int n = wn * 64 + j * 16 + l15;
#pragma unroll
            for (int h = 0; h < 16; ++h)
                bf[j][h] = Bs[cur][n * LDSS + fragk(lane, h)];
        }
#pragma unroll
        for (int i = 0; i < 2; ++i)
#pragma unroll
            for (int j = 0; j < 4; ++j)
                acc[i][j] = __builtin_amdgcn_wmma_f32_16x16x32_bf16(
                    false, af[i], false, bf[j], (short)0, acc[i][j], false, false);

        if (has_next) storeTile(cur ^ 1);         // fill ping-pong buffer
        __syncthreads();
        cur ^= 1;
    }

#pragma unroll
    for (int i = 0; i < 2; ++i)
#pragma unroll
        for (int j = 0; j < 4; ++j)
#pragma unroll
            for (int r = 0; r < 8; ++r) {
                const int rr = bm + wm * 32 + i * 16 + r + hi * 8;
                const int cc = bn + wn * 64 + j * 16 + l15;
                C[(size_t)rr * N + cc] = acc[i][j][r];
            }
}

// ---------------------------------------------------------------------------
// Post: RMSNorm + RoPE on q/k heads, plain convert for v. One block per
// (slot, q, b); slot 0..15 = q-head, 16..19 = k-head, 20..23 = v-head.
// Outputs bf16, head-major [b][h][q][d].
// ---------------------------------------------------------------------------
__global__ __launch_bounds__(128)
void qkv_post(const float* __restrict__ qkv, const int* __restrict__ positions,
              const float* __restrict__ qw, const float* __restrict__ kw,
              unsigned short* __restrict__ qb, unsigned short* __restrict__ kb,
              unsigned short* __restrict__ vb)
{
    const int slot = blockIdx.x, q = blockIdx.y, b = blockIdx.z;
    const int d = threadIdx.x;                       // 0..127
    const size_t rowoff = (size_t)(b * Q_ + q) * QKVN;

    if (slot >= 20) {                                // V head: straight copy
        const int hv = slot - 20;
        float x = qkv[rowoff + HQ_*D_ + HKV_*D_ + hv * D_ + d];
        vb[(((size_t)b * HKV_ + hv) * Q_ + q) * D_ + d] =
            __builtin_bit_cast(unsigned short, f2bf(x));
        return;
    }

    __shared__ float red[128];
    __shared__ float xs[128];

    const bool isq = slot < 16;
    const int off  = isq ? slot * D_ : HQ_*D_ + (slot - 16) * D_;
    float x = qkv[rowoff + off + d];

    red[d] = x * x;
    __syncthreads();
#pragma unroll
    for (int s = 64; s > 0; s >>= 1) {
        if (d < s) red[d] += red[d + s];
        __syncthreads();
    }
    const float rms = rsqrtf(red[0] * (1.0f / D_) + 1e-6f);
    const float w   = isq ? qw[d] : kw[d];
    xs[d] = x * rms * w;
    __syncthreads();

    const int pos = positions[b * Q_ + q];
    const int j   = d & 63;
    const float inv  = __expf(-(float)j * (13.815510557964274f / 64.0f)); // theta=1e6
    const float fr   = (float)pos * inv;
    const float c    = __cosf(fr), sn = __sinf(fr);
    float out = (d < 64) ? (xs[d] * c - xs[d + 64] * sn)
                         : (xs[d] * c + xs[d - 64] * sn);

    if (isq)
        qb[(((size_t)b * HQ_ + slot) * Q_ + q) * D_ + d] =
            __builtin_bit_cast(unsigned short, f2bf(out));
    else
        kb[(((size_t)b * HKV_ + (slot - 16)) * Q_ + q) * D_ + d] =
            __builtin_bit_cast(unsigned short, f2bf(out));
}

// ---------------------------------------------------------------------------
// Flash attention, causal, GQA (4 q-heads share one kv-head).
// Block = (b, h, 64-row q tile); 4 waves, each owns 16 q rows.
// K/V chunks (64 x 128 bf16) double-buffered in LDS; staged with
// GLOBAL_LOAD_ASYNC_TO_LDS_B128 when available (ASYNCcnt), else via VGPRs.
// ---------------------------------------------------------------------------
__global__ __launch_bounds__(128)
void attn_fa(const unsigned short* __restrict__ qb,
             const unsigned short* __restrict__ kb,
             const unsigned short* __restrict__ vb,
             unsigned short* __restrict__ attn_out)
{
    const int KV_STRIDE = 144;           // 128 + 16 pad (halves) -> 16B aligned rows
    const int P_STRIDE  = 72;            // 64 + 8 pad
    __shared__ __attribute__((aligned(16))) __bf16 Ks[2][64 * KV_STRIDE];
    __shared__ __attribute__((aligned(16))) __bf16 Vs[2][64 * KV_STRIDE];
    __shared__ __bf16 Ps[4][16 * P_STRIDE];

    const int qt  = blockIdx.x;          // q tile (64 rows)
    const int h   = blockIdx.y;
    const int b   = blockIdx.z;
    const int hk  = h >> 2;              // GQA: HQ/HKV = 4
    const int tid = threadIdx.x;
    const int wid = tid >> 5, lane = tid & 31;
    const int l15 = lane & 15, hi = (lane >> 4) & 1;
    const int q0  = qt * 64 + wid * 16;  // wave's global q base
    const float scale = 0.08838834764831845f;   // 1/sqrt(128)

    const unsigned short* qbase = qb + (((size_t)b * HQ_  + h ) * Q_) * D_;
    const unsigned short* kbase = kb + (((size_t)b * HKV_ + hk) * Q_) * D_;
    const unsigned short* vbase = vb + (((size_t)b * HKV_ + hk) * Q_) * D_;

    const int row  = tid >> 1;           // 0..63
    const int colb = (tid & 1) * 64;     // 0 or 64 (halves)

    // stage one 64x128 K/V chunk into LDS buffer `buf`
    auto stage = [&](int ck, int buf) {
        const uint4* ksrc = (const uint4*)(kbase + (size_t)(ck * 64 + row) * D_ + colb);
        const uint4* vsrc = (const uint4*)(vbase + (size_t)(ck * 64 + row) * D_ + colb);
        uint4* kdst = (uint4*)(&Ks[buf][row * KV_STRIDE + colb]);
        uint4* vdst = (uint4*)(&Vs[buf][row * KV_STRIDE + colb]);
#if HAS_ASYNC_LDS
#pragma unroll
        for (int c = 0; c < 8; ++c) {
            __builtin_amdgcn_global_load_async_to_lds_b128(
                (__attribute__((address_space(1))) i4vs*)(ksrc + c),
                (__attribute__((address_space(3))) i4vs*)(kdst + c), 0, 0);
            __builtin_amdgcn_global_load_async_to_lds_b128(
                (__attribute__((address_space(1))) i4vs*)(vsrc + c),
                (__attribute__((address_space(3))) i4vs*)(vdst + c), 0, 0);
        }
#else
#pragma unroll
        for (int c = 0; c < 8; ++c) { kdst[c] = ksrc[c]; vdst[c] = vsrc[c]; }
#endif
    };

    // Load Q fragments once: 4 chunks of K=32 over D=128
    v16bf qf[4];
#pragma unroll
    for (int kc = 0; kc < 4; ++kc)
#pragma unroll
        for (int t = 0; t < 16; ++t)
            qf[kc][t] = __builtin_bit_cast(__bf16,
                qbase[(size_t)(q0 + l15) * D_ + kc * 32 + fragk(lane, t)]);

    v8f o[8];
#pragma unroll
    for (int dt = 0; dt < 8; ++dt) o[dt] = (v8f){};
    float mrow[8], lrow[8];
#pragma unroll
    for (int r = 0; r < 8; ++r) { mrow[r] = -1e30f; lrow[r] = 0.0f; }

    stage(0, 0);
    wait_async_lds();
    __syncthreads();

    for (int ck = 0; ck <= qt; ++ck) {
        const int cur = ck & 1;
        if (ck < qt) stage(ck + 1, cur ^ 1);       // prefetch next chunk (async)

        // S = Q K^T over D (contraction)
        v8f s[4];
#pragma unroll
        for (int nt = 0; nt < 4; ++nt) {
            s[nt] = (v8f){};
#pragma unroll
            for (int kc = 0; kc < 4; ++kc) {
                v16bf bk;
#pragma unroll
                for (int t = 0; t < 16; ++t)
                    bk[t] = Ks[cur][(nt * 16 + l15) * KV_STRIDE + kc * 32 + fragk(lane, t)];
                s[nt] = __builtin_amdgcn_wmma_f32_16x16x32_bf16(
                    false, qf[kc], false, bk, (short)0, s[nt], false, false);
            }
        }

        // scale + causal mask (only diagonal chunk needs per-element mask)
        const bool diag = (ck == qt);
#pragma unroll
        for (int nt = 0; nt < 4; ++nt)
#pragma unroll
            for (int r = 0; r < 8; ++r) {
                float v = s[nt][r] * scale;
                if (diag) {
                    const int qg = q0 + r + hi * 8;
                    const int kv = ck * 64 + nt * 16 + l15;
                    if (kv > qg) v = -1e30f;
                }
                s[nt][r] = v;
            }

        // online softmax; row stats via 16-lane shfl-xor reductions
#pragma unroll
        for (int r = 0; r < 8; ++r) {
            float mx = s[0][r];
#pragma unroll
            for (int nt = 1; nt < 4; ++nt) mx = fmaxf(mx, s[nt][r]);
#pragma unroll
            for (int off = 1; off <= 8; off <<= 1)
                mx = fmaxf(mx, __shfl_xor(mx, off, 32));
            const float mnew  = fmaxf(mrow[r], mx);
            const float alpha = __expf(mrow[r] - mnew);
            mrow[r] = mnew;

            float lloc = 0.0f;
#pragma unroll
            for (int nt = 0; nt < 4; ++nt) {
                const float p = __expf(s[nt][r] - mnew);
                Ps[wid][(r + hi * 8) * P_STRIDE + nt * 16 + l15] = f2bf(p);
                lloc += p;
            }
#pragma unroll
            for (int off = 1; off <= 8; off <<= 1)
                lloc += __shfl_xor(lloc, off, 32);
            lrow[r] = lrow[r] * alpha + lloc;
#pragma unroll
            for (int dt = 0; dt < 8; ++dt) o[dt][r] *= alpha;
        }

        // O += P V  (contraction over 64 kv, 2 chunks of 32)
        v16bf pa[2];
#pragma unroll
        for (int kc = 0; kc < 2; ++kc)
#pragma unroll
            for (int t = 0; t < 16; ++t)
                pa[kc][t] = Ps[wid][l15 * P_STRIDE + kc * 32 + fragk(lane, t)];
#pragma unroll
        for (int dt = 0; dt < 8; ++dt)
#pragma unroll
            for (int kc = 0; kc < 2; ++kc) {
                v16bf bv;
#pragma unroll
                for (int t = 0; t < 16; ++t)
                    bv[t] = Vs[cur][(kc * 32 + fragk(lane, t)) * KV_STRIDE + dt * 16 + l15];
                o[dt] = __builtin_amdgcn_wmma_f32_16x16x32_bf16(
                    false, pa[kc], false, bv, (short)0, o[dt], false, false);
            }

        wait_async_lds();     // next chunk fully in LDS
        __syncthreads();      // all waves done with cur before it is reused
    }

    // epilogue: O / l -> bf16 [b*Q + q][h*D + d]
#pragma unroll
    for (int dt = 0; dt < 8; ++dt)
#pragma unroll
        for (int r = 0; r < 8; ++r) {
            const int   qg  = q0 + r + hi * 8;
            const float val = o[dt][r] / lrow[r];
            attn_out[(size_t)(b * Q_ + qg) * (HQ_ * D_) + h * D_ + dt * 16 + l15] =
                __builtin_bit_cast(unsigned short, f2bf(val));
        }
}

// ---------------------------------------------------------------------------
// kernel_launch
// inputs: 0=positions(i32) 1=hidden(f32) 2=k_cache 3=v_cache 4=wqkv(f32)
//         5=wo(f32) 6=q_norm_w 7=k_norm_w 8=seq_len
// ---------------------------------------------------------------------------
extern "C" void kernel_launch(void* const* d_in, const int* in_sizes, int n_in,
                              void* d_out, int out_size, void* d_ws, size_t ws_size,
                              hipStream_t stream)
{
    const int*   positions = (const int*)  d_in[0];
    const float* hidden    = (const float*)d_in[1];
    const float* wqkv      = (const float*)d_in[4];
    const float* wo        = (const float*)d_in[5];
    const float* qw        = (const float*)d_in[6];
    const float* kw        = (const float*)d_in[7];
    float*       out       = (float*)d_out;

    char* ws = (char*)d_ws;
    float*          qkv   = (float*)ws;           ws += (size_t)NTOK * QKVN * 4;   // 48 MB
    unsigned short* qbuf  = (unsigned short*)ws;  ws += (size_t)B_*HQ_*Q_*D_ * 2;  // 16 MB
    unsigned short* kbuf  = (unsigned short*)ws;  ws += (size_t)B_*HKV_*Q_*D_ * 2; //  4 MB
    unsigned short* vbuf  = (unsigned short*)ws;  ws += (size_t)B_*HKV_*Q_*D_ * 2; //  4 MB
    unsigned short* attnb = (unsigned short*)ws;                                    // 16 MB

    // 1) QKV projection: [4096 x 3072] = hidden [4096 x 2048] @ wqkv^T
    gemm_wmma_bf16<float><<<dim3(QKVN / 128, NTOK / 128), 256, 0, stream>>>(
        hidden, wqkv, qkv, NTOK, QKVN, HID_);

    // 2) RMSNorm + RoPE + layout, bf16
    qkv_post<<<dim3(HQ_ + 2 * HKV_, Q_, B_), 128, 0, stream>>>(
        qkv, positions, qw, kw, qbuf, kbuf, vbuf);

    // 3) causal flash attention
    attn_fa<<<dim3(Q_ / 64, HQ_, B_), 128, 0, stream>>>(qbuf, kbuf, vbuf, attnb);

    // 4) output projection: [4096 x 2048] = attn [4096 x 2048] @ wo^T
    gemm_wmma_bf16<unsigned short><<<dim3(HID_ / 128, NTOK / 128), 256, 0, stream>>>(
        attnb, wo, out, NTOK, HID_, HQ_ * D_);
}
